// GRUCell_10952166604829
// MI455X (gfx1250) — matrix operand order repaired and checked by reading, not compiled, not run on hardware
//
#include <hip/hip_runtime.h>
#include <cstdint>
#include <cstddef>

// GRU cell: B=8192, D_IN=1024, D_H=1024, fp32 in/out.
// Pass 0: convert x,h -> bf16; convert+transpose all 6 weights -> bf16 [n][k].
// Kernel 1: z/r gates + xn via 5 fused GEMMs (bf16 WMMA, f32 accum).
// Kernel 2: (r*h) @ Uhn + output combine.
// Staging: activations via global_load_async_to_lds_b128 (ASYNCcnt),
//          weight tiles via TDM tensor_load_to_lds (TENSORcnt, wave 0 issues),
//          double-buffered LDS, one barrier per K-step.

#define NB 8192
#define DK 1024

typedef __attribute__((ext_vector_type(16))) __bf16 v16bf;
typedef __attribute__((ext_vector_type(8)))  float  v8f;
typedef __attribute__((ext_vector_type(4)))  __bf16 v4bf;
typedef __attribute__((ext_vector_type(4)))  unsigned int v4u;
typedef __attribute__((ext_vector_type(8)))  unsigned int v8u;

__device__ __forceinline__ float sigm(float v) {
    return 1.0f / (1.0f + __expf(-v));
}

__device__ __forceinline__ unsigned short bf_bits(float f) {
    __bf16 b = (__bf16)f;
    return __builtin_bit_cast(unsigned short, b);
}

__device__ __forceinline__ v4bf pk4(float4 v) {
    v4bf r; r.x = (__bf16)v.x; r.y = (__bf16)v.y; r.z = (__bf16)v.z; r.w = (__bf16)v.w; return r;
}

__device__ __forceinline__ void async_b128(unsigned lds_off, const void* gaddr) {
    asm volatile("global_load_async_to_lds_b128 %0, %1, off"
                 :: "v"(lds_off), "v"(gaddr) : "memory");
}
__device__ __forceinline__ void wait_async0() {
    asm volatile("s_wait_asynccnt 0x0" ::: "memory");
}

// ---------------------------------------------------------------------------
// TDM: load a 2D tile (rows_tile x 32 bf16, tensor row stride DK elements)
// into LDS with 16B padding after each 64B row (-> [rows][40] bf16 layout).
// D# group0: count=1 | lds_addr | global_addr | type=2
// D# group1: data_size=2B, pad_enable, pad_interval=16dw, pad_amount=4dw,
//            tensor_dim0=DK, tensor_dim1=rows_tensor, tile_dim0=32,
//            tile_dim1=rows_tile, tensor_dim0_stride=DK.
// ---------------------------------------------------------------------------
__device__ __forceinline__ void tdm_load_2d(unsigned lds_off, const void* gaddr,
                                            unsigned rows_tile, unsigned rows_tensor) {
    unsigned long long ga = (unsigned long long)(uintptr_t)gaddr;
    v4u g0;
    g0.x = 1u;                                                  // count=1, user desc
    g0.y = lds_off;                                             // lds_addr
    g0.z = (unsigned)ga;                                        // global_addr[31:0]
    g0.w = (unsigned)((ga >> 32) & 0x1FFFFFFu) | (2u << 30);    // addr[56:32] | type=2
    v8u g1;
    g1.s0 = (1u << 16) | (1u << 20) | (3u << 22) | (3u << 25);  // 2B elems, pad 16dw -> +4dw
    g1.s1 = (unsigned)DK << 16;                                 // tensor_dim0[15:0]
    g1.s2 = rows_tensor << 16;                                  // tensor_dim1[15:0]
    g1.s3 = 32u << 16;                                          // tile_dim0 = 32
    g1.s4 = rows_tile;                                          // tile_dim1
    g1.s5 = (unsigned)DK;                                       // tensor_dim0_stride[31:0]
    g1.s6 = 0u;
    g1.s7 = 0u;
    asm volatile("tensor_load_to_lds %0, %1" :: "s"(g0), "s"(g1) : "memory");
}

// ---------------------------------------------------------------------------
// Pass 0a: elementwise f32 -> bf16 (x, h). Exact cover: 2048 blk * 256 thr * 4 it.
// ---------------------------------------------------------------------------
__global__ __launch_bounds__(256) void cvt_bf16_kernel(const float* __restrict__ src,
                                                       unsigned short* __restrict__ dst) {
    const size_t stride = (size_t)gridDim.x * 256;
    size_t i = (size_t)blockIdx.x * 256 + threadIdx.x;
    #pragma unroll
    for (int it = 0; it < 4; ++it) {
        float4 v = *(const float4*)&src[i * 4];
        *(v4bf*)&dst[i * 4] = pk4(v);
        i += stride;
    }
}

// ---------------------------------------------------------------------------
// Pass 0b: weight convert + transpose: Wt[n*DK + k] = bf16(W[k*DK + n]).
// 64x64 tiles via LDS; grid (16,16,6).
// ---------------------------------------------------------------------------
struct WtArgs {
    const float* src[6];
    unsigned short* dst[6];
};

__global__ __launch_bounds__(256) void wt_cvt_kernel(WtArgs args) {
    __shared__ __align__(16) __bf16 t[64][72];
    const int tid = threadIdx.x;
    const int k0 = blockIdx.x * 64;
    const int n0 = blockIdx.y * 64;
    const float* __restrict__ W = args.src[blockIdx.z];
    unsigned short* __restrict__ Wt = args.dst[blockIdx.z];

    #pragma unroll
    for (int it = 0; it < 4; ++it) {
        int c  = tid + it * 256;
        int r  = c >> 4;
        int c4 = (c & 15) << 2;
        float4 v = *(const float4*)&W[(size_t)(k0 + r) * DK + n0 + c4];
        t[c4+0][r] = (__bf16)v.x; t[c4+1][r] = (__bf16)v.y;
        t[c4+2][r] = (__bf16)v.z; t[c4+3][r] = (__bf16)v.w;
    }
    __syncthreads();
    #pragma unroll
    for (int it = 0; it < 2; ++it) {
        int c  = tid + it * 256;
        int r  = c >> 3;
        int o8 = (c & 7) << 3;
        *(uint4*)&Wt[(size_t)(n0 + r) * DK + k0 + o8] = *(const uint4*)&t[r][o8];
    }
}

// ---------------------------------------------------------------------------
// Kernel 1: 128x64 tile; accz = x@Wiz + h@Uhz, accr = x@Wir + h@Uhr, accn = x@Win.
// ---------------------------------------------------------------------------
__global__ __launch_bounds__(256) void gru_gates_kernel(
    const unsigned short* __restrict__ xbf, const unsigned short* __restrict__ hbf,
    const unsigned short* __restrict__ wtz, const unsigned short* __restrict__ utz,
    const unsigned short* __restrict__ wtr, const unsigned short* __restrict__ utr,
    const unsigned short* __restrict__ wtn,
    const float* __restrict__ bz, const float* __restrict__ br,
    const float* __restrict__ h,
    float* __restrict__ zbuf, float* __restrict__ xnbuf, unsigned short* __restrict__ rhbuf)
{
    constexpr int BM = 128, BK = 32, LP = 40;

    __shared__ __align__(16) __bf16 xs[2][BM][LP];
    __shared__ __align__(16) __bf16 hs[2][BM][LP];
    __shared__ __align__(16) __bf16 ws_[2][5][64][LP];

    const int tid  = threadIdx.x;
    const int wave = tid >> 5;
    const int lane = tid & 31;
    const int m0 = blockIdx.x * BM;
    const int n0 = blockIdx.y * 64;

    const unsigned short* Wt[5] = {wtz, utz, wtr, utr, wtn};

    const int arow = wave * 16 + (lane & 15);
    const int akh  = (lane >> 4) << 3;
    const int bcol = lane & 15;
    const int bkh  = (lane >> 4) << 4;

    const int ar_ = tid >> 2;
    const int ao8 = (tid & 3) << 3;

    v8f accz[4], accr[4], accn[4];
    const v8f vzero = {0.f,0.f,0.f,0.f,0.f,0.f,0.f,0.f};
    #pragma unroll
    for (int t = 0; t < 4; ++t) { accz[t] = vzero; accr[t] = vzero; accn[t] = vzero; }

    auto stage = [&](int k0, int pb) {
        // x/h activation tiles: per-lane async 16B chunks
        #pragma unroll
        for (int it = 0; it < 2; ++it) {
            int r = ar_ + it * 64;
            async_b128((unsigned)(uintptr_t)&xs[pb][r][ao8],
                       &xbf[(size_t)(m0 + r) * DK + k0 + ao8]);
            async_b128((unsigned)(uintptr_t)&hs[pb][r][ao8],
                       &hbf[(size_t)(m0 + r) * DK + k0 + ao8]);
        }
        // weight tiles: one TDM descriptor each, issued by wave 0
        if (wave == 0) {
            #pragma unroll
            for (int w = 0; w < 5; ++w) {
                tdm_load_2d((unsigned)(uintptr_t)&ws_[pb][w][0][0],
                            &Wt[w][(size_t)n0 * DK + k0], 64u, (unsigned)DK);
            }
        }
    };

    stage(0, 0);
    wait_async0();
    if (wave == 0) __builtin_amdgcn_s_wait_tensorcnt(0);

    int pb = 0;
    for (int k0 = 0; k0 < DK; k0 += BK) {
        const int nk = k0 + BK;
        __syncthreads();
        if (nk < DK) stage(nk, pb ^ 1);

        v16bf xa, ha;
        #pragma unroll
        for (int i = 0; i < 8; ++i) {
            xa[i]   = xs[pb][arow][bkh + i];
            xa[i+8] = xs[pb][arow][bkh + 8 + i];
            ha[i]   = hs[pb][arow][bkh + i];
            ha[i+8] = hs[pb][arow][bkh + 8 + i];
        }
        #pragma unroll
        for (int nt = 0; nt < 4; ++nt) {
            const int bn_ = nt * 16 + bcol;
            v16bf f0, f1, f2, f3, f4;
            #pragma unroll
            for (int i = 0; i < 16; ++i) {
                f0[i] = ws_[pb][0][bn_][bkh + i];
                f1[i] = ws_[pb][1][bn_][bkh + i];
                f2[i] = ws_[pb][2][bn_][bkh + i];
                f3[i] = ws_[pb][3][bn_][bkh + i];
                f4[i] = ws_[pb][4][bn_][bkh + i];
            }
            accz[nt] = __builtin_amdgcn_wmma_f32_16x16x32_bf16(false, xa, false, f0, (short)0, accz[nt], false, false);
            accz[nt] = __builtin_amdgcn_wmma_f32_16x16x32_bf16(false, ha, false, f1, (short)0, accz[nt], false, false);
            accr[nt] = __builtin_amdgcn_wmma_f32_16x16x32_bf16(false, xa, false, f2, (short)0, accr[nt], false, false);
            accr[nt] = __builtin_amdgcn_wmma_f32_16x16x32_bf16(false, ha, false, f3, (short)0, accr[nt], false, false);
            accn[nt] = __builtin_amdgcn_wmma_f32_16x16x32_bf16(false, xa, false, f4, (short)0, accn[nt], false, false);
        }

        wait_async0();                                  // own async x/h chunks done
        if (wave == 0) __builtin_amdgcn_s_wait_tensorcnt(0);  // TDM weight tiles done
        pb ^= 1;
    }

    const int em = m0 + wave * 16 + ((lane >> 4) << 3);
    #pragma unroll
    for (int nt = 0; nt < 4; ++nt) {
        const int n = n0 + nt * 16 + bcol;
        const float bzv = bz[n];
        const float brv = br[n];
        #pragma unroll
        for (int j = 0; j < 8; ++j) {
            const size_t idx = (size_t)(em + j) * DK + n;
            float zz = sigm(accz[nt][j] + bzv);
            float rr = sigm(accr[nt][j] + brv);
            float hv = h[idx];
            zbuf[idx]  = zz;
            xnbuf[idx] = accn[nt][j];
            rhbuf[idx] = bf_bits(rr * hv);
        }
    }
}

// ---------------------------------------------------------------------------
// Kernel 2: acc = (r*h) @ Uhn ; n = sigm(xn + acc + bn); out = (1-z)*h + z*n.
// ---------------------------------------------------------------------------
__global__ __launch_bounds__(256) void gru_out_kernel(
    const unsigned short* __restrict__ rh, const unsigned short* __restrict__ unt,
    const float* __restrict__ bn, const float* __restrict__ xnbuf,
    const float* __restrict__ zbuf, const float* __restrict__ h,
    float* __restrict__ out)
{
    constexpr int BM = 128, BK = 32, LP = 40;

    __shared__ __align__(16) __bf16 as_[2][BM][LP];
    __shared__ __align__(16) __bf16 uns[2][64][LP];

    const int tid  = threadIdx.x;
    const int wave = tid >> 5;
    const int lane = tid & 31;
    const int m0 = blockIdx.x * BM;
    const int n0 = blockIdx.y * 64;

    const int arow = wave * 16 + (lane & 15);
    const int akh  = (lane >> 4) << 3;
    const int bcol = lane & 15;
    const int bkh  = (lane >> 4) << 4;

    const int ar_ = tid >> 2;
    const int ao8 = (tid & 3) << 3;

    v8f acc[4];
    const v8f vzero = {0.f,0.f,0.f,0.f,0.f,0.f,0.f,0.f};
    #pragma unroll
    for (int t = 0; t < 4; ++t) acc[t] = vzero;

    auto stage = [&](int k0, int pb) {
        #pragma unroll
        for (int it = 0; it < 2; ++it) {
            int r = ar_ + it * 64;
            async_b128((unsigned)(uintptr_t)&as_[pb][r][ao8],
                       &rh[(size_t)(m0 + r) * DK + k0 + ao8]);
        }
        if (wave == 0) {
            tdm_load_2d((unsigned)(uintptr_t)&uns[pb][0][0],
                        &unt[(size_t)n0 * DK + k0], 64u, (unsigned)DK);
        }
    };

    stage(0, 0);
    wait_async0();
    if (wave == 0) __builtin_amdgcn_s_wait_tensorcnt(0);

    int pb = 0;
    for (int k0 = 0; k0 < DK; k0 += BK) {
        const int nk = k0 + BK;
        __syncthreads();
        if (nk < DK) stage(nk, pb ^ 1);

        v16bf aa;
        #pragma unroll
        for (int i = 0; i < 8; ++i) {
            aa[i]   = as_[pb][arow][bkh + i];
            aa[i+8] = as_[pb][arow][bkh + 8 + i];
        }
        #pragma unroll
        for (int nt = 0; nt < 4; ++nt) {
            const int bn_ = nt * 16 + bcol;
            v16bf fb;
            #pragma unroll
            for (int i = 0; i < 16; ++i) fb[i] = uns[pb][bn_][bkh + i];
            acc[nt] = __builtin_amdgcn_wmma_f32_16x16x32_bf16(false, aa, false, fb, (short)0, acc[nt], false, false);
        }

        wait_async0();
        if (wave == 0) __builtin_amdgcn_s_wait_tensorcnt(0);
        pb ^= 1;
    }

    const int em = m0 + wave * 16 + ((lane >> 4) << 3);
    #pragma unroll
    for (int nt = 0; nt < 4; ++nt) {
        const int n = n0 + nt * 16 + bcol;
        const float bnv = bn[n];
        #pragma unroll
        for (int j = 0; j < 8; ++j) {
            const size_t idx = (size_t)(em + j) * DK + n;
            float nv = sigm(xnbuf[idx] + acc[nt][j] + bnv);
            float zv = zbuf[idx];
            float hv = h[idx];
            out[idx] = (1.0f - zv) * hv + zv * nv;
        }
    }
}

// ---------------------------------------------------------------------------
extern "C" void kernel_launch(void* const* d_in, const int* in_sizes, int n_in,
                              void* d_out, int out_size, void* d_ws, size_t ws_size,
                              hipStream_t stream) {
    (void)in_sizes; (void)n_in; (void)out_size; (void)ws_size;
    const float* x   = (const float*)d_in[0];
    const float* h   = (const float*)d_in[1];
    const float* Wiz = (const float*)d_in[2];
    const float* Uhz = (const float*)d_in[3];
    const float* bz  = (const float*)d_in[4];
    const float* Wir = (const float*)d_in[5];
    const float* Uhr = (const float*)d_in[6];
    const float* br  = (const float*)d_in[7];
    const float* Win = (const float*)d_in[8];
    const float* Uhn = (const float*)d_in[9];
    const float* bn  = (const float*)d_in[10];
    float* out = (float*)d_out;

    const size_t elems = (size_t)NB * DK;       // 8M
    const size_t welems = (size_t)DK * DK;      // 1M
    char* ws = (char*)d_ws;
    float* zbuf  = (float*)ws;                               ws += elems * 4;
    float* xnbuf = (float*)ws;                               ws += elems * 4;
    unsigned short* rhbuf = (unsigned short*)ws;             ws += elems * 2;
    unsigned short* xbf   = (unsigned short*)ws;             ws += elems * 2;
    unsigned short* hbf   = (unsigned short*)ws;             ws += elems * 2;
    unsigned short* wt[6];
    for (int w = 0; w < 6; ++w) { wt[w] = (unsigned short*)ws; ws += welems * 2; }

    cvt_bf16_kernel<<<2048, 256, 0, stream>>>(x, xbf);
    cvt_bf16_kernel<<<2048, 256, 0, stream>>>(h, hbf);
    WtArgs wa;
    wa.src[0] = Wiz; wa.src[1] = Uhz; wa.src[2] = Wir;
    wa.src[3] = Uhr; wa.src[4] = Win; wa.src[5] = Uhn;
    for (int w = 0; w < 6; ++w) wa.dst[w] = wt[w];
    wt_cvt_kernel<<<dim3(16, 16, 6), 256, 0, stream>>>(wa);

    dim3 grid(NB / 128, DK / 64);
    dim3 blk(256);
    gru_gates_kernel<<<grid, blk, 0, stream>>>(xbf, hbf, wt[0], wt[1], wt[2], wt[3], wt[4],
                                               bz, br, h, zbuf, xnbuf, rhbuf);
    gru_out_kernel<<<grid, blk, 0, stream>>>(rhbuf, wt[5], bn, xnbuf, zbuf, h, out);
}
